// LatentLanguage_22359599743508
// MI455X (gfx1250) — matrix operand order repaired and checked by reading, compiled, not run on hardware
//
#include <hip/hip_runtime.h>
#include <hip/hip_bf16.h>

#define B_ 2
#define L_ 2048
#define T_ (B_*L_)
#define DIM_ 1024
#define VOCAB_ 32000
#define NL_ 4
#define DSTATE_ 128
#define DCONV_ 4
#define HEADDIM_ 64
#define DIN_ 2048
#define H_ 32
#define CONVDIM_ (DIN_ + 2*DSTATE_)          /* 2304 */
#define DPROJ_ (2*DIN_ + 2*DSTATE_ + H_)     /* 4384 */
#define FFN_ 4096
#define EPS_ 1e-6f

typedef __attribute__((ext_vector_type(16))) __bf16 v16bf;
typedef __attribute__((ext_vector_type(2)))  __bf16 v2bf;
typedef __attribute__((ext_vector_type(8)))  float  v8f;

// Exact pointee type of the async-to-LDS builtins (per hipcc diagnostic):
// int __attribute__((vector_size(16))) in AS1 (global) / AS3 (LDS).
typedef int i32x4_vs __attribute__((vector_size(16)));
typedef __attribute__((address_space(1))) i32x4_vs g_i32x4;
typedef __attribute__((address_space(3))) i32x4_vs l_i32x4;

// gfx1250 async global->LDS path (ASYNCcnt-tracked); fall back if toolchain
// lacks the builtins (fallback also taken in the host pass — harmless).
#if __has_builtin(__builtin_amdgcn_global_load_async_to_lds_b128) && \
    __has_builtin(__builtin_amdgcn_s_wait_asynccnt)
#define HAVE_ASYNC_LDS 1
#define ASYNC_WAIT() __builtin_amdgcn_s_wait_asynccnt(0)
#else
#define HAVE_ASYNC_LDS 0
#define ASYNC_WAIT()
#endif

__device__ __forceinline__ g_i32x4* to_gas(const void* p) {
    return (g_i32x4*)(uintptr_t)p;            // generic -> global AS reinterpret
}
__device__ __forceinline__ l_i32x4* to_lds(void* p) {
    return (l_i32x4*)(unsigned)(uintptr_t)p;  // generic LDS ptr low 32 bits == LDS offset
}

// Packed f32->bf16 pair via native fptrunc (backend selects the hardware
// convert if gfx1250 has one; otherwise emits the RNE software sequence).
__device__ __forceinline__ unsigned pack2bf(float lo, float hi) {
    v2bf v;
    v.x = (__bf16)lo;
    v.y = (__bf16)hi;
    return __builtin_bit_cast(unsigned, v);
}

__device__ __forceinline__ float silu_(float x) { return x / (1.f + __expf(-x)); }

struct U8x32 { uint4 lo, hi; };   // 32 bytes == v16bf

// ---------------------------------------------------------------------------
// Tiled GEMM: C[M,N] = A[M,K] @ W[N,K]^T (+ residual or bias), bf16 WMMA.
// 128x128 block tile, BK=32, 256 threads = 8 waves (4x2), wave tile 32x64.
// Staging: float4 global loads -> packed bf16 -> uint4 LDS stores.
// Fragments: 16B-aligned ds_load_b128 pairs.
// EPI: 0 = none, 1 = C = X[M,N] + AW^T, 2 = C = AW^T + X[N] (bias)
// ---------------------------------------------------------------------------
template<int EPI>
__global__ void __launch_bounds__(256) gemm_kernel(
    const float* __restrict__ A, const float* __restrict__ W,
    const float* __restrict__ X, float* __restrict__ C,
    int M, int N, int K)
{
    __shared__ alignas(16) unsigned short sA[128][40];  // bf16, 80B row stride
    __shared__ alignas(16) unsigned short sW[128][40];

    const int tid  = threadIdx.x;
    const int lane = tid & 31;
    const int wave = tid >> 5;
    const int m0 = blockIdx.y * 128, n0 = blockIdx.x * 128;
    const int wm = (wave & 3) * 32;          // wave M offset in block tile
    const int wn = (wave >> 2) * 64;         // wave N offset in block tile
    const int half = lane >> 4, l16 = lane & 15;

    v8f acc[2][4];
    #pragma unroll
    for (int i = 0; i < 2; ++i)
        #pragma unroll
        for (int j = 0; j < 4; ++j)
            #pragma unroll
            for (int r = 0; r < 8; ++r) acc[i][j][r] = 0.f;

    for (int k0 = 0; k0 < K; k0 += 32) {
        __syncthreads();
        // Stage 128x32 tiles: each thread handles 2 chunks of 8 floats.
        #pragma unroll
        for (int j = 0; j < 2; ++j) {
            int ch = tid + 256 * j;                 // 0..511
            int r  = ch >> 2, ks = (ch & 3) * 8;    // row, k-segment
            // A tile
            {
                int gm = m0 + r;
                float4 a0 = make_float4(0.f, 0.f, 0.f, 0.f), a1 = a0;
                if (gm < M) {
                    const float* p = &A[(size_t)gm * K + k0 + ks];
                    a0 = *(const float4*)p;
                    a1 = *(const float4*)(p + 4);
                }
                uint4 pk;
                pk.x = pack2bf(a0.x, a0.y); pk.y = pack2bf(a0.z, a0.w);
                pk.z = pack2bf(a1.x, a1.y); pk.w = pack2bf(a1.z, a1.w);
                *(uint4*)&sA[r][ks] = pk;
            }
            // W tile
            {
                int gn = n0 + r;
                float4 w0 = make_float4(0.f, 0.f, 0.f, 0.f), w1 = w0;
                if (gn < N) {
                    const float* p = &W[(size_t)gn * K + k0 + ks];
                    w0 = *(const float4*)p;
                    w1 = *(const float4*)(p + 4);
                }
                uint4 pk;
                pk.x = pack2bf(w0.x, w0.y); pk.y = pack2bf(w0.z, w0.w);
                pk.z = pack2bf(w1.x, w1.y); pk.w = pack2bf(w1.z, w1.w);
                *(uint4*)&sW[r][ks] = pk;
            }
        }
        // Prefetch next K-tile (global_prefetch_b8).
        if (k0 + 32 < K) {
            int gm = m0 + (tid >> 1);
            if (gm < M) __builtin_prefetch(&A[(size_t)gm * K + k0 + 32 + (tid & 1) * 16], 0, 0);
            int gn = n0 + (tid >> 1);
            if (gn < N) __builtin_prefetch(&W[(size_t)gn * K + k0 + 32 + (tid & 1) * 16], 0, 0);
        }
        __syncthreads();

        // A fragments: lane half -> k in {0..7,16..23} or {8..15,24..31}
        v16bf afrag[2];
        #pragma unroll
        for (int mf = 0; mf < 2; ++mf) {
            const unsigned short* row = &sA[wm + mf * 16 + l16][0];
            U8x32 u;
            u.lo = *(const uint4*)(row + 8 * half);
            u.hi = *(const uint4*)(row + 16 + 8 * half);
            afrag[mf] = __builtin_bit_cast(v16bf, u);
        }
        // B fragments: lane half -> k in {0..7,16..23} or {8..15,24..31}
        #pragma unroll
        for (int nf = 0; nf < 4; ++nf) {
            const unsigned short* row = &sW[wn + nf * 16 + l16][0];
            U8x32 u;
            u.lo = *(const uint4*)(row + 8 * half);
            u.hi = *(const uint4*)(row + 16 + 8 * half);
            v16bf bfrag = __builtin_bit_cast(v16bf, u);
            #pragma unroll
            for (int mf = 0; mf < 2; ++mf) {
                acc[mf][nf] = __builtin_amdgcn_wmma_f32_16x16x32_bf16(
                    false, afrag[mf], false, bfrag, (short)0, acc[mf][nf],
                    false, false);
            }
        }
    }

    // Epilogue: C layout — lane<16: n=lane, m=r; lane>=16: n=lane-16, m=8+r.
    #pragma unroll
    for (int mf = 0; mf < 2; ++mf) {
        #pragma unroll
        for (int nf = 0; nf < 4; ++nf) {
            int n = n0 + wn + nf * 16 + l16;
            if (n >= N) continue;
            #pragma unroll
            for (int r = 0; r < 8; ++r) {
                int m = m0 + wm + mf * 16 + 8 * half + r;
                if (m >= M) continue;
                float v = acc[mf][nf][r];
                if (EPI == 1) v += X[(size_t)m * N + n];
                if (EPI == 2) v += X[n];
                C[(size_t)m * N + n] = v;
            }
        }
    }
}

// ---------------------------------------------------------------------------
// Embedding gather (float4)
// ---------------------------------------------------------------------------
__global__ void __launch_bounds__(256) embed_kernel(
    const int* __restrict__ tok, const float* __restrict__ emb, float* __restrict__ x)
{
    int t = blockIdx.x;
    int id = tok[t];
    int d4 = threadIdx.x;                      // DIM/4 = 256
    ((float4*)&x[(size_t)t * DIM_])[d4] = ((const float4*)&emb[(size_t)id * DIM_])[d4];
}

// ---------------------------------------------------------------------------
// RMSNorm: out = in*(silu(z) optional) * rsqrt(mean(sq)+eps) * (w optional)
// ---------------------------------------------------------------------------
__global__ void __launch_bounds__(256) rmsnorm_kernel(
    const float* __restrict__ in, int in_stride,
    const float* __restrict__ z, int z_stride,
    const float* __restrict__ w,
    float* __restrict__ out, int n)
{
    __shared__ float red[256];
    int row = blockIdx.x, tid = threadIdx.x;
    const float* ir = in + (size_t)row * in_stride;
    const float* zr = z ? z + (size_t)row * z_stride : nullptr;
    float ssq = 0.f;
    for (int d = tid; d < n; d += 256) {
        float v = ir[d];
        if (zr) v *= silu_(zr[d]);
        ssq += v * v;
    }
    red[tid] = ssq;
    __syncthreads();
    for (int s = 128; s > 0; s >>= 1) {
        if (tid < s) red[tid] += red[tid + s];
        __syncthreads();
    }
    float scale = rsqrtf(red[0] / (float)n + EPS_);
    for (int d = tid; d < n; d += 256) {
        float v = ir[d];
        if (zr) v *= silu_(zr[d]);
        v *= scale;
        if (w) v *= w[d];
        out[(size_t)row * n + d] = v;
    }
}

// ---------------------------------------------------------------------------
// Causal depthwise conv (DCONV=4) + bias + silu on the xBC slice of zxbcdt.
// ---------------------------------------------------------------------------
__global__ void __launch_bounds__(256) conv_silu_kernel(
    const float* __restrict__ zx, const float* __restrict__ cw,
    const float* __restrict__ cb, float* __restrict__ out)
{
    size_t i = (size_t)blockIdx.x * 256 + threadIdx.x;
    if (i >= (size_t)T_ * CONVDIM_) return;
    int c = (int)(i % CONVDIM_);
    int t = (int)(i / CONVDIM_);
    int b = t / L_, tl = t % L_;
    float acc = cb[c];
    #pragma unroll
    for (int k = 0; k < DCONV_; ++k) {
        int tt = tl - (DCONV_ - 1) + k;
        if (tt >= 0)
            acc += cw[c * DCONV_ + k] * zx[((size_t)(b * L_ + tt)) * DPROJ_ + DIN_ + c];
    }
    out[i] = silu_(acc);
}

// ---------------------------------------------------------------------------
// dt = softplus(dtraw + dt_bias)
// ---------------------------------------------------------------------------
__global__ void __launch_bounds__(256) dt_kernel(
    const float* __restrict__ zx, const float* __restrict__ dtb, float* __restrict__ dt)
{
    int i = blockIdx.x * 256 + threadIdx.x;
    if (i >= T_ * H_) return;
    int t = i / H_, h = i % H_;
    float v = zx[(size_t)t * DPROJ_ + DIN_ + CONVDIM_ + h] + dtb[h];
    dt[i] = (v > 20.f) ? v : log1pf(__expf(v));
}

// ---------------------------------------------------------------------------
// Per-timestep staging of B(128f) + C(128f) + x_head(64f) = 1280 B into LDS.
// Async path: 80 lanes each issue one GLOBAL_LOAD_ASYNC_TO_LDS_B128.
// ---------------------------------------------------------------------------
__device__ __forceinline__ void ssd_stage_issue(
    const float* __restrict__ xBCc, float* dst, size_t rowx, int h, int tid)
{
    if (tid < 80) {
        size_t g;
        if (tid < 32)      g = rowx + DIN_ + (size_t)tid * 4;
        else if (tid < 64) g = rowx + DIN_ + DSTATE_ + (size_t)(tid - 32) * 4;
        else               g = rowx + (size_t)h * HEADDIM_ + (size_t)(tid - 64) * 4;
#if HAVE_ASYNC_LDS
        __builtin_amdgcn_global_load_async_to_lds_b128(
            to_gas(xBCc + g), to_lds(dst + tid * 4), 0, 0);
#else
        *(float4*)(dst + tid * 4) = *(const float4*)(xBCc + g);
#endif
    }
}

// ---------------------------------------------------------------------------
// SSD as an exact sequential recurrence per (batch, head):
//   S = exp(dt*A)*S + dt*B⊗x ;  y = C·S + D*x
// 256 threads: p = tid&63, n-group = tid>>6 (32 n-values in registers).
// Double-buffered async prefetch of step t+1 during step t.
// ---------------------------------------------------------------------------
__global__ void __launch_bounds__(256) ssd_scan_kernel(
    const float* __restrict__ xBCc, const float* __restrict__ dt,
    const float* __restrict__ A_log, const float* __restrict__ Dp,
    float* __restrict__ y)
{
    int bh = blockIdx.x;
    int b = bh / H_, h = bh % H_;
    int tid = threadIdx.x, p = tid & 63, ng = tid >> 6;
    __shared__ alignas(16) float stage[2][2 * DSTATE_ + HEADDIM_];  // B | C | x
    __shared__ float ypart[4][HEADDIM_];
    float Ah = -__expf(A_log[h]);
    float Dh = Dp[h];
    float s[32];
    #pragma unroll
    for (int j = 0; j < 32; ++j) s[j] = 0.f;

    ssd_stage_issue(xBCc, &stage[0][0], (size_t)(b * L_) * CONVDIM_, h, tid);

    int cur = 0;
    for (int t = 0; t < L_; ++t) {
        ASYNC_WAIT();
        __syncthreads();
        if (t + 1 < L_)
            ssd_stage_issue(xBCc, &stage[cur ^ 1][0],
                            (size_t)(b * L_ + t + 1) * CONVDIM_, h, tid);

        float dtv = dt[(size_t)(b * L_ + t) * H_ + h];
        float da  = __expf(dtv * Ah);
        const float* Bp = &stage[cur][ng * 32];
        const float* Cp = &stage[cur][DSTATE_ + ng * 32];
        float dtx  = dtv * stage[cur][2 * DSTATE_ + p];
        float yacc = 0.f;
        #pragma unroll
        for (int j = 0; j < 32; ++j) {
            s[j] = s[j] * da + Bp[j] * dtx;
            yacc += Cp[j] * s[j];
        }
        ypart[ng][p] = yacc;
        __syncthreads();
        if (tid < HEADDIM_) {
            float yv = ypart[0][tid] + ypart[1][tid] + ypart[2][tid] + ypart[3][tid]
                     + Dh * stage[cur][2 * DSTATE_ + tid];
            y[(size_t)(b * L_ + t) * DIN_ + h * HEADDIM_ + tid] = yv;
        }
        __syncthreads();
        cur ^= 1;
    }
}

// ---------------------------------------------------------------------------
// Elementwise gate: u *= v
// ---------------------------------------------------------------------------
__global__ void __launch_bounds__(256) mul_kernel(
    float* __restrict__ u, const float* __restrict__ v, size_t n)
{
    size_t i = (size_t)blockIdx.x * 256 + threadIdx.x;
    if (i < n) u[i] *= v[i];
}

// ---------------------------------------------------------------------------
extern "C" void kernel_launch(void* const* d_in, const int* in_sizes, int n_in,
                              void* d_out, int out_size, void* d_ws, size_t ws_size,
                              hipStream_t stream)
{
    (void)in_sizes; (void)n_in; (void)out_size; (void)ws_size;

    const int*   tokens    = (const int*)  d_in[0];
    const float* embedding = (const float*)d_in[1];
    const float* lm_bias   = (const float*)d_in[2];
    const float* in_proj   = (const float*)d_in[3];
    const float* conv_w    = (const float*)d_in[4];
    const float* conv_b    = (const float*)d_in[5];
    const float* dt_bias   = (const float*)d_in[6];
    const float* A_log     = (const float*)d_in[7];
    const float* Dp        = (const float*)d_in[8];
    const float* mnorm_w   = (const float*)d_in[9];
    const float* out_proj  = (const float*)d_in[10];
    const float* u_w       = (const float*)d_in[11];
    const float* v_w       = (const float*)d_in[12];
    const float* o_w       = (const float*)d_in[13];
    float* out = (float*)d_out;

    // Workspace layout (floats), lifetime-based aliasing:
    //   regA: zxbcdt (T*DPROJ) then FFN u (T*FFN)
    //   regB: conv out (T*CONVDIM) then yn (T*DIN) then FFN v (T*FFN)
    float* ws  = (float*)d_ws;
    size_t off = 0;
    float* x    = ws + off; off += (size_t)T_ * DIM_;
    float* xn   = ws + off; off += (size_t)T_ * DIM_;
    float* regA = ws + off; off += (size_t)T_ * DPROJ_;
    float* regB = ws + off; off += (size_t)T_ * FFN_;
    float* y    = ws + off; off += (size_t)T_ * DIN_;
    float* dt   = ws + off; off += (size_t)T_ * H_;
    // total ~207 MB

    dim3 blk(256);
    embed_kernel<<<T_, blk, 0, stream>>>(tokens, embedding, x);

    for (int li = 0; li < NL_; ++li) {
        const float* Wi  = in_proj  + (size_t)li * DPROJ_ * DIM_;
        const float* cw  = conv_w   + (size_t)li * CONVDIM_ * DCONV_;
        const float* cb  = conv_b   + (size_t)li * CONVDIM_;
        const float* db  = dt_bias  + (size_t)li * H_;
        const float* Al  = A_log    + (size_t)li * H_;
        const float* Dl  = Dp       + (size_t)li * H_;
        const float* nw  = mnorm_w  + (size_t)li * DIN_;
        const float* Wo  = out_proj + (size_t)li * DIM_ * DIN_;
        const float* Wu  = u_w      + (size_t)li * FFN_ * DIM_;
        const float* Wv  = v_w      + (size_t)li * FFN_ * DIM_;
        const float* Woo = o_w      + (size_t)li * DIM_ * FFN_;

        for (int it = 0; it < 2; ++it) {
            rmsnorm_kernel<<<T_, blk, 0, stream>>>(x, DIM_, nullptr, 0, nullptr, xn, DIM_);
            gemm_kernel<0><<<dim3((DPROJ_ + 127) / 128, T_ / 128), blk, 0, stream>>>(
                xn, Wi, nullptr, regA, T_, DPROJ_, DIM_);
            {
                size_t nel = (size_t)T_ * CONVDIM_;
                conv_silu_kernel<<<(unsigned)((nel + 255) / 256), blk, 0, stream>>>(
                    regA, cw, cb, regB);
            }
            dt_kernel<<<(T_ * H_ + 255) / 256, blk, 0, stream>>>(regA, db, dt);
            ssd_scan_kernel<<<B_ * H_, blk, 0, stream>>>(regB, dt, Al, Dl, y);
            rmsnorm_kernel<<<T_, blk, 0, stream>>>(y, DIN_, regA, DPROJ_, nw, regB, DIN_);
            gemm_kernel<1><<<dim3(DIM_ / 128, T_ / 128), blk, 0, stream>>>(
                regB, Wo, xn, x, T_, DIM_, DIN_);

            rmsnorm_kernel<<<T_, blk, 0, stream>>>(x, DIM_, nullptr, 0, nullptr, xn, DIM_);
            gemm_kernel<0><<<dim3(FFN_ / 128, T_ / 128), blk, 0, stream>>>(
                xn, Wu, nullptr, regA, T_, FFN_, DIM_);
            gemm_kernel<0><<<dim3(FFN_ / 128, T_ / 128), blk, 0, stream>>>(
                xn, Wv, nullptr, regB, T_, FFN_, DIM_);
            mul_kernel<<<(unsigned)(((size_t)T_ * FFN_ + 255) / 256), blk, 0, stream>>>(
                regA, regB, (size_t)T_ * FFN_);
            gemm_kernel<1><<<dim3(DIM_ / 128, T_ / 128), blk, 0, stream>>>(
                regA, Woo, xn, x, T_, DIM_, FFN_);
        }
    }

    gemm_kernel<2><<<dim3((VOCAB_ + 127) / 128, T_ / 128), blk, 0, stream>>>(
        x, embedding, lm_bias, out, T_, VOCAB_, DIM_);
}
